// BackwardInjectConv2D_63617055588734
// MI455X (gfx1250) — compile-verified, matrix operand order checked by reading, not run on hardware
//
#include <hip/hip_runtime.h>

// ---------------------------------------------------------------------------
// Backward of stride-2 3x3 conv, NHWC, fp32 in/out.
//   grad_output: [32,28,28,256]  d_in[0]
//   inputs     : [32,56,56,256]  d_in[1]
//   kernels    : [3,3,256,256]   d_in[2]
// Outputs concatenated in d_out:
//   grad_in  [32,56,56,256] @ 0
//   grad_wt  [3,3,256,256]  @ 25690112
//   grad_bias[256]          @ 26279936
// Strategy: bf16 WMMA (v_wmma_f32_16x16x32_bf16) with f32 accumulation.
// ---------------------------------------------------------------------------

typedef __attribute__((ext_vector_type(16))) __bf16 v16bf;
typedef __attribute__((ext_vector_type(8)))  float  v8f;

union FragBF { v16bf v; unsigned short u[16]; };

__device__ __forceinline__ unsigned short f32_bf16(float f) {
  union { float f; unsigned u; } a; a.f = f;
  unsigned r = a.u + 0x7FFFu + ((a.u >> 16) & 1u);   // round-to-nearest-even
  return (unsigned short)(r >> 16);
}

__device__ __forceinline__ void cvt8(FragBF& A, int o, const float4& a, const float4& b) {
  A.u[o+0] = f32_bf16(a.x); A.u[o+1] = f32_bf16(a.y);
  A.u[o+2] = f32_bf16(a.z); A.u[o+3] = f32_bf16(a.w);
  A.u[o+4] = f32_bf16(b.x); A.u[o+5] = f32_bf16(b.y);
  A.u[o+6] = f32_bf16(b.z); A.u[o+7] = f32_bf16(b.w);
}

// ---------------------------------------------------------------------------
// 1) Pack rotated/transposed weights Wt[tap=(wr,wc)][co][ci] = kernels[wr][wc][ci][co]
//    into per-lane WMMA B-fragment order, bf16.
//    Packed half index: (((tap*8+kb)*16+nt)*32+lane)*16 + idx
//    lane = (n&15) | (((k>>3)&1)<<4), idx = (k&7) | ((k>>4)<<3), k=co%32, n=ci%16
// ---------------------------------------------------------------------------
__global__ void k_pack_weights(const float* __restrict__ kern,
                               unsigned short* __restrict__ wpack) {
  int gid = blockIdx.x * 256 + threadIdx.x;            // < 9*8*16*32*16 = 589824
  int idx  = gid & 15;
  int lane = (gid >> 4) & 31;
  int nt   = (gid >> 9) & 15;
  int kb   = (gid >> 13) & 7;
  int tap  = gid >> 16;                                // 0..8
  if (tap >= 9) return;
  int sel = lane >> 4, n = lane & 15;
  int k = (idx < 8) ? (sel * 8 + idx) : (16 + sel * 8 + (idx - 8));
  int co = kb * 32 + k;
  int ci = nt * 16 + n;
  float v = kern[((long)tap * 256 + ci) * 256 + co];
  wpack[gid] = f32_bf16(v);
}

__global__ void k_zero(float* __restrict__ p, int n) {
  int i = blockIdx.x * 256 + threadIdx.x;
  if (i < n) p[i] = 0.f;
}

// ---------------------------------------------------------------------------
// 2) grad_in[b,y,x,ci] = sum over taps (y+ky even, x+kx even) of
//      grad_output[b,(y+ky-2)/2,(x+kx-2)/2,:] . kernels[2-ky,2-kx,ci,:]
//    One wave per tile = (b, y, x-parity, 16-pixel block) x 256 ci.
//    M=16 pixels, N=256 ci (16 WMMA n-tiles), K=256 co per tap (1..4 taps).
// ---------------------------------------------------------------------------
__global__ __launch_bounds__(256) void k_grad_in(const float* __restrict__ go,
                                                 const v16bf* __restrict__ wp,
                                                 float* __restrict__ gi) {
  int w    = threadIdx.x >> 5;
  int lane = threadIdx.x & 31;
  int tile = blockIdx.x * 8 + w;                       // 7168 tiles total
  int tblk = tile & 1;
  int px   = (tile >> 1) & 1;
  int y    = (tile >> 2) % 56;
  int b    = tile / 224;                               // 4*56
  int sel  = lane >> 4, lm = lane & 15;
  int t0   = tblk * 16;

  // row taps: weight row wr = 2-ky, input row i
  int wrs[2], is_[2]; int nrow = 0;
  if (y & 1) { wrs[0] = 1; is_[0] = (y - 1) >> 1; nrow = 1; }
  else {
    if (y >= 2) { wrs[nrow] = 2; is_[nrow] = (y - 2) >> 1; nrow++; }
    wrs[nrow] = 0; is_[nrow] = y >> 1; nrow++;         // y/2 <= 27 for even y
  }
  // col taps: weight col wc = 2-kx, j = t + dj
  int wcs[2], dj[2]; int ncol;
  if (px) { wcs[0] = 1; dj[0] = 0; ncol = 1; }
  else    { wcs[0] = 2; dj[0] = -1; wcs[1] = 0; dj[1] = 0; ncol = 2; }

  v8f C[16];
  #pragma unroll
  for (int q = 0; q < 16; q++) {
    #pragma unroll
    for (int e = 0; e < 8; e++) C[q][e] = 0.f;
  }

  for (int rt = 0; rt < nrow; ++rt)
    for (int ct = 0; ct < ncol; ++ct) {
      int i = is_[rt];
      int j = t0 + lm + dj[ct];                        // this lane's A row pixel
      bool jv = (j >= 0) && (j < 28);
      const float* arow = go + (((long)(b * 28 + i)) * 28 + (jv ? j : 0)) * 256;
      int tap = wrs[rt] * 3 + wcs[ct];
      const v16bf* wt = wp + (long)tap * 8 * 16 * 32;
      #pragma unroll
      for (int kb = 0; kb < 8; ++kb) {
        FragBF A;
        int c0 = kb * 32 + sel * 8;
        float4 f0, f1, f2, f3;
        if (jv) {
          f0 = *(const float4*)(arow + c0);
          f1 = *(const float4*)(arow + c0 + 4);
          f2 = *(const float4*)(arow + c0 + 16);
          f3 = *(const float4*)(arow + c0 + 20);
        } else {
          f0 = make_float4(0.f, 0.f, 0.f, 0.f); f1 = f0; f2 = f0; f3 = f0;
        }
        cvt8(A, 0, f0, f1);
        cvt8(A, 8, f2, f3);
        const v16bf* wkb = wt + (long)kb * 16 * 32 + lane;
        #pragma unroll
        for (int nt = 0; nt < 16; ++nt) {
          v16bf Bf = wkb[nt * 32];
          C[nt] = __builtin_amdgcn_wmma_f32_16x16x32_bf16(
              false, A.v, false, Bf, (short)0, C[nt], false, false);
        }
      }
    }

  // store: C lane layout: col = lane&15 (ci), row = v + 8*(lane>>4) (pixel)
  #pragma unroll
  for (int nt = 0; nt < 16; ++nt) {
    #pragma unroll
    for (int v = 0; v < 8; ++v) {
      int mrow = v + 8 * sel;
      int t = t0 + mrow;
      if (t < 28) {
        int x = px + 2 * t;
        gi[(((long)(b * 56 + y)) * 56 + x) * 256 + nt * 16 + lm] = C[nt][v];
      }
    }
  }
}

// ---------------------------------------------------------------------------
// 3) grad_wt[ky,kx,ci,co] = sum_{b,i,j} inputs[b,ky+2i,kx+2j,ci] * go[b,i,j,co]
//    (zero when ky+2i>=56 or kx+2j>=56). GEMM: M=ci, N=co, K=25088 pixels.
//    WG tile 128ci x 128co, K split 16 ways, 32-pixel K-steps staged through
//    LDS in WMMA fragment order; accumulate with global f32 atomics.
// ---------------------------------------------------------------------------
__global__ __launch_bounds__(256) void k_grad_wt(const float* __restrict__ go,
                                                 const float* __restrict__ inp,
                                                 float* __restrict__ gw) {
  __shared__ __align__(32) unsigned short ldsA[8 * 32 * 16];  // 8 m-tiles
  __shared__ __align__(32) unsigned short ldsB[8 * 32 * 16];  // 8 n-tiles
  int bx  = blockIdx.x;                                // 9*2*2*16 = 576
  int kc  = bx & 15;
  int cob = (bx >> 4) & 1;
  int cib = (bx >> 5) & 1;
  int tap = bx >> 6;                                   // 0..8
  int ky = tap / 3, kx = tap % 3;
  int tid = threadIdx.x;
  int lane = tid & 31, w = tid >> 5;
  int sel = lane >> 4, lm = lane & 15;

  v8f C[8];
  #pragma unroll
  for (int q = 0; q < 8; q++) {
    #pragma unroll
    for (int e = 0; e < 8; e++) C[q][e] = 0.f;
  }

  for (int it = 0; it < 49; ++it) {                    // 16*49*32 = 25088 pixels
    int pbase = kc * 1568 + it * 32;
    // stage 128x32 (A: inputs ci-major) and 32x128 (B: grad_output co) tiles
    for (int e = tid; e < 4096; e += 256) {
      int cl = e & 127, pl = e >> 7;
      int p  = pbase + pl;
      int bb = p / 784; int r = p - bb * 784;
      int ii = r / 28;  int jj = r - ii * 28;
      int row = ky + 2 * ii, col = kx + 2 * jj;
      float av = (row < 56 && col < 56)
          ? inp[(((long)(bb * 56 + row)) * 56 + col) * 256 + cib * 128 + cl] : 0.f;
      float bv = go[((long)p) * 256 + cob * 128 + cl];
      int lA = (cl & 15) | (((pl >> 3) & 1) << 4);     // fragment lane
      int ix = (pl & 7) | ((pl >> 4) << 3);            // fragment half-index
      int t16 = cl >> 4;                               // 16-wide sub-tile
      ldsA[(t16 * 32 + lA) * 16 + ix] = f32_bf16(av);
      ldsB[(t16 * 32 + lA) * 16 + ix] = f32_bf16(bv);
    }
    __syncthreads();
    v16bf Af = *(const v16bf*)&ldsA[(w * 32 + lane) * 16];
    #pragma unroll
    for (int nt = 0; nt < 8; ++nt) {
      v16bf Bf = *(const v16bf*)&ldsB[(nt * 32 + lane) * 16];
      C[nt] = __builtin_amdgcn_wmma_f32_16x16x32_bf16(
          false, Af, false, Bf, (short)0, C[nt], false, false);
    }
    __syncthreads();
  }

  #pragma unroll
  for (int nt = 0; nt < 8; ++nt) {
    #pragma unroll
    for (int v = 0; v < 8; ++v) {
      int ci = cib * 128 + w * 16 + v + 8 * sel;
      int co = cob * 128 + nt * 16 + lm;
      atomicAdd(&gw[((long)tap * 256 + ci) * 256 + co], C[nt][v]);
    }
  }
}

// ---------------------------------------------------------------------------
// 4) grad_bias[c] = sum_{b,i,j} grad_output[b,i,j,c]
// ---------------------------------------------------------------------------
__global__ void k_grad_bias(const float* __restrict__ go, float* __restrict__ gb) {
  int tid = threadIdx.x;                               // channel
  const float* base = go + (long)blockIdx.x * 256 * 256;  // 98 pixel chunks
  float s = 0.f;
  for (int p = 0; p < 256; ++p) s += base[p * 256 + tid];
  atomicAdd(&gb[tid], s);
}

// ---------------------------------------------------------------------------
extern "C" void kernel_launch(void* const* d_in, const int* in_sizes, int n_in,
                              void* d_out, int out_size, void* d_ws, size_t ws_size,
                              hipStream_t stream) {
  const float* go   = (const float*)d_in[0];
  const float* inp  = (const float*)d_in[1];
  const float* kern = (const float*)d_in[2];
  float* out = (float*)d_out;

  const long base_wt   = 32L * 56 * 56 * 256;          // 25690112
  const long base_bias = base_wt + 9L * 256 * 256;     // 26279936
  const int  tail_n    = 9 * 256 * 256 + 256;          // 590080

  unsigned short* wpack = (unsigned short*)d_ws;       // 1.125 MB bf16 packed weights

  k_pack_weights<<<2304, 256, 0, stream>>>(kern, wpack);
  k_zero<<<(tail_n + 255) / 256, 256, 0, stream>>>(out + base_wt, tail_n);
  k_grad_in<<<896, 256, 0, stream>>>(go, (const v16bf*)d_ws, out);
  k_grad_wt<<<576, 256, 0, stream>>>(go, inp, out + base_wt);
  k_grad_bias<<<98, 256, 0, stream>>>(go, out + base_bias);
}
